// BinaryThinkingNet_24696061952678
// MI455X (gfx1250) — compile-verified, matrix operand order checked by reading, not run on hardware
//
#include <hip/hip_runtime.h>
#include <hip/hip_bf16.h>
#include <math.h>

// Problem constants (match reference)
#define BB    4
#define TT    4096
#define DDIM  1024
#define HHH   16
#define DHH   64
#define CCH   64
#define DFFF  2816
#define LLL   4
#define NCH   (TT / CCH)          // 64 chunks
#define MTOT  (BB * TT)           // 16384 rows
#define EPSF  1e-6f
#define SCALEF 0.125f             // 1/sqrt(64)

typedef __attribute__((ext_vector_type(16))) __bf16 v16bf;
typedef __attribute__((ext_vector_type(8)))  __bf16 v8bf;
typedef __attribute__((ext_vector_type(8)))  float  v8f;
typedef __attribute__((ext_vector_type(4)))  unsigned int uint32x4;
typedef __attribute__((ext_vector_type(8)))  int int32x8;
typedef __attribute__((ext_vector_type(4)))  int int32x4;

union V16 { v16bf v; v8bf h[2]; };

#if __has_builtin(__builtin_amdgcn_tensor_load_to_lds)
#define HAVE_TDM 1
#else
#define HAVE_TDM 0
#endif

// K-offset map for 16-bit WMMA A/B fragments (ISA 7.12.2):
// lane half hh = lane>>4; elems 0..7 -> K = hh*8 + j ; elems 8..15 -> K = 16 + hh*8 + (j-8)
__device__ __forceinline__ int kmap(int j, int hh) {
  return (j < 8) ? (hh * 8 + j) : (16 + hh * 8 + (j - 8));
}

// Row-major fragment: row = row0 + (lane&15); two contiguous 8-elem (16B) segments.
__device__ __forceinline__ v16bf frag_row(const __bf16* base, int row0, int ld, int kb, int lane) {
  int r  = row0 + (lane & 15);
  int hh = (lane >> 4) & 1;
  const __bf16* p = base + (size_t)r * ld + kb + hh * 8;
  V16 u;
  u.h[0] = *(const v8bf*)(p);
  u.h[1] = *(const v8bf*)(p + 16);
  return u.v;
}

// Column gather from LDS bf16 tile (stride 64): transposed-A or transposed-B fragment.
__device__ __forceinline__ v16bf frag_col_bf16(const __bf16* s, int kb, int n0, int lane) {
  int n  = n0 + (lane & 15);
  int hh = (lane >> 4) & 1;
  V16 u;
#pragma unroll
  for (int j = 0; j < 16; ++j)
    u.v[j] = s[(kb + kmap(j, hh)) * 64 + n];
  return u.v;
}

// Same gather from fp32 LDS (running attention state), converting to bf16.
__device__ __forceinline__ v16bf frag_col_f32(const float* s, int kb, int n0, int lane) {
  int n  = n0 + (lane & 15);
  int hh = (lane >> 4) & 1;
  V16 u;
#pragma unroll
  for (int j = 0; j < 16; ++j)
    u.v[j] = (__bf16)s[(kb + kmap(j, hh)) * 64 + n];
  return u.v;
}

__device__ __forceinline__ v8f wmma_bf16(v16bf a, v16bf b, v8f c) {
  return __builtin_amdgcn_wmma_f32_16x16x32_bf16(false, a, false, b, (short)0, c, false, false);
}

#if HAVE_TDM
// LDS byte offset of a static __shared__ object (generic -> AS(3) strips aperture)
__device__ __forceinline__ unsigned lds_off_of(const void* p) {
  return (unsigned)(size_t)(__attribute__((address_space(3))) const char*)p;
}

// Issue a 2-D TDM tile load: tile_k x tile_rows bf16 elements, row stride = stride_elems,
// packed row-major into LDS at lds_addr. Descriptor bit layout per CDNA5 ISA ch.8.
__device__ __forceinline__ void tdm_load_2d(unsigned lds_addr, const void* gptr,
                                            unsigned tile_k, unsigned tile_rows,
                                            unsigned tensor_k, unsigned tensor_rows,
                                            unsigned stride_elems) {
  unsigned long long ga = (unsigned long long)(size_t)gptr;
  uint32x4 g0;
  g0[0] = 1u;                                              // count=1, user descriptor
  g0[1] = lds_addr;                                        // LDS byte address
  g0[2] = (unsigned)(ga & 0xFFFFFFFFu);                    // global_addr[31:0]
  g0[3] = (unsigned)((ga >> 32) & 0x01FFFFFFu) | (2u << 30); // addr[56:32] | type=2
  unsigned long long s0 = stride_elems;                    // dim0 stride (48-bit field)
  int32x8 g1;
  g1[0] = (int)(1u << 16);                                 // data_size=1 (2 bytes)
  g1[1] = (int)((tensor_k & 0xFFFFu) << 16);               // tensor_dim0[15:0] @ bits 63:48
  g1[2] = (int)((tensor_k >> 16) | ((tensor_rows & 0xFFFFu) << 16)); // dim0 hi | dim1 lo
  g1[3] = (int)((tensor_rows >> 16) | (tile_k << 16));     // dim1 hi | tile_dim0
  g1[4] = (int)(tile_rows & 0xFFFFu);                      // tile_dim1 (tile_dim2 = 0)
  g1[5] = (int)(s0 & 0xFFFFFFFFu);                         // stride0[31:0]
  g1[6] = (int)((s0 >> 32) & 0xFFFFu);                     // stride0[47:32] | stride1 lo = 0
  g1[7] = 0;
  int32x4 z4 = {0, 0, 0, 0};
#if defined(__clang_major__) && (__clang_major__ >= 23)
  int32x8 z8 = {0, 0, 0, 0, 0, 0, 0, 0};
  __builtin_amdgcn_tensor_load_to_lds(g0, g1, z4, z4, z8, 0);
#else
  __builtin_amdgcn_tensor_load_to_lds(g0, g1, z4, z4, 0);
#endif
}
#endif  // HAVE_TDM

// ---------------------------------------------------------------------------
// Embedding: x[b,t,d] = sum_i bits(byte)[i] * bit_in_w[d,i] + pos_emb[t,d]
// ---------------------------------------------------------------------------
__global__ void embed_kernel(const int* __restrict__ ids, const float* __restrict__ bw,
                             const float* __restrict__ pe, float* __restrict__ X) {
  size_t idx = (size_t)blockIdx.x * 256 + threadIdx.x;   // over MTOT*DDIM
  int d = (int)(idx % DDIM);
  size_t bt = idx / DDIM;
  int t = (int)(bt % TT);
  int byte = ids[bt];
  float s = pe[(size_t)t * DDIM + d];
#pragma unroll
  for (int i = 0; i < 8; ++i)
    if ((byte >> (7 - i)) & 1) s += bw[d * 8 + i];
  X[idx] = s;
}

// ---------------------------------------------------------------------------
// RMSNorm fp32 row -> bf16 row. One 256-thread block per row of 1024.
// ---------------------------------------------------------------------------
__global__ void __launch_bounds__(256) rmsnorm_kernel(const float* __restrict__ X,
                                                      const float* __restrict__ wn,
                                                      __bf16* __restrict__ Y) {
  __shared__ float red[8];
  size_t rowo = (size_t)blockIdx.x * DDIM;
  int tid = threadIdx.x;
  int d = tid * 4;
  float4 xv = *(const float4*)(X + rowo + d);
  float ss = xv.x * xv.x + xv.y * xv.y + xv.z * xv.z + xv.w * xv.w;
#pragma unroll
  for (int o = 16; o > 0; o >>= 1) ss += __shfl_xor(ss, o, 32);
  if ((tid & 31) == 0) red[tid >> 5] = ss;
  __syncthreads();
  float tot = red[0] + red[1] + red[2] + red[3] + red[4] + red[5] + red[6] + red[7];
  float inv = rsqrtf(tot / (float)DDIM + EPSF);
  Y[rowo + d + 0] = (__bf16)(xv.x * inv * wn[d + 0]);
  Y[rowo + d + 1] = (__bf16)(xv.y * inv * wn[d + 1]);
  Y[rowo + d + 2] = (__bf16)(xv.z * inv * wn[d + 2]);
  Y[rowo + d + 3] = (__bf16)(xv.w * inv * wn[d + 3]);
}

// ---------------------------------------------------------------------------
// TDM-staged WMMA GEMM: Out[m,n] = sum_k A[m,k] * W[n,k]  (A: MxK, W: NxK, bf16)
// Block 256 thr (8 waves), tile 128x64, K-tile 64, double-buffered LDS fed by
// tensor_load_to_lds (TENSORcnt). Wave tile 32x32 = 2x2 WMMA accumulators.
// Epilogues: 0 plain bf16, 1 *alpha bf16, 2 binarize bf16, 3 residual-add fp32
// ---------------------------------------------------------------------------
enum { EPI_BF16 = 0, EPI_SCALE = 1, EPI_BIN = 2, EPI_RESID = 3 };

template <int EPI>
__global__ void __launch_bounds__(256) gemm_bf16_wmma(const __bf16* __restrict__ A,
                                                      const __bf16* __restrict__ W,
                                                      const float* resid, float* outF,
                                                      __bf16* outB, int M, int N, int K,
                                                      float alpha) {
  __shared__ __bf16 sA[2][128 * 64];   // 2 x 16 KB
  __shared__ __bf16 sW[2][64 * 64];    // 2 x  8 KB

  int lane = threadIdx.x & 31;
  int w    = threadIdx.x >> 5;
  int mblk = blockIdx.y * 128;
  int nblk = blockIdx.x * 64;
  int m0   = (w >> 1) * 32;            // within block tile
  int n0   = (w & 1)  * 32;

  int nt = K / 64;

#if HAVE_TDM
  if (w == 0) {
    tdm_load_2d(lds_off_of(&sA[0][0]), A + (size_t)mblk * K, 64, 128, K, M, K);
    tdm_load_2d(lds_off_of(&sW[0][0]), W + (size_t)nblk * K, 64, 64,  K, N, K);
  }
#endif

  v8f acc00 = {}, acc01 = {}, acc10 = {}, acc11 = {};

  for (int t = 0; t < nt; ++t) {
#if HAVE_TDM
    int cur = t & 1;
    if (w == 0) {
      if (t + 1 < nt) {
        int nb = (t + 1) & 1;
        tdm_load_2d(lds_off_of(&sA[nb][0]), A + (size_t)mblk * K + (t + 1) * 64,
                    64, 128, K, M, K);
        tdm_load_2d(lds_off_of(&sW[nb][0]), W + (size_t)nblk * K + (t + 1) * 64,
                    64, 64,  K, N, K);
        __builtin_amdgcn_s_wait_tensorcnt(2);   // current tile's two DMAs done
      } else {
        __builtin_amdgcn_s_wait_tensorcnt(0);
      }
    }
    __syncthreads();                            // publish LDS tile to all waves
#else
    int cur = 0;
    __syncthreads();
    {
      int tid = threadIdx.x;
      const __bf16* gA = A + (size_t)mblk * K + t * 64;
      const __bf16* gW = W + (size_t)nblk * K + t * 64;
      for (int i = tid; i < 128 * 8; i += 256) {
        int r = i >> 3, c = (i & 7) * 8;
        *(v8bf*)(&sA[0][r * 64 + c]) = *(const v8bf*)(gA + (size_t)r * K + c);
      }
      for (int i = tid; i < 64 * 8; i += 256) {
        int r = i >> 3, c = (i & 7) * 8;
        *(v8bf*)(&sW[0][r * 64 + c]) = *(const v8bf*)(gW + (size_t)r * K + c);
      }
    }
    __syncthreads();
#endif

    const __bf16* a_ = &sA[cur][0];
    const __bf16* w_ = &sW[cur][0];
#pragma unroll
    for (int kb = 0; kb < 64; kb += 32) {
      v16bf a0 = frag_row(a_, m0,      64, kb, lane);
      v16bf a1 = frag_row(a_, m0 + 16, 64, kb, lane);
      v16bf b0 = frag_row(w_, n0,      64, kb, lane);
      v16bf b1 = frag_row(w_, n0 + 16, 64, kb, lane);
      acc00 = wmma_bf16(a0, b0, acc00);
      acc01 = wmma_bf16(a0, b1, acc01);
      acc10 = wmma_bf16(a1, b0, acc10);
      acc11 = wmma_bf16(a1, b1, acc11);
    }
    __syncthreads();                            // tile fully consumed before reuse
  }

  int hh = lane >> 4, nn = lane & 15;
  v8f* accs[4] = {&acc00, &acc01, &acc10, &acc11};
#pragma unroll
  for (int i = 0; i < 2; ++i) {
#pragma unroll
    for (int j = 0; j < 2; ++j) {
      v8f a = *accs[i * 2 + j];
#pragma unroll
      for (int r = 0; r < 8; ++r) {
        int m = mblk + m0 + i * 16 + r + 8 * hh;
        int n = nblk + n0 + j * 16 + nn;
        size_t o = (size_t)m * N + n;
        float v = a[r];
        if (EPI == EPI_BF16)       outB[o] = (__bf16)v;
        else if (EPI == EPI_SCALE) outB[o] = (__bf16)(v * alpha);
        else if (EPI == EPI_BIN)   outB[o] = (__bf16)((v >= 0.0f) ? 1.0f : -1.0f);
        else                       outF[o] = resid[o] + v;
      }
    }
  }
}

// ---------------------------------------------------------------------------
// Chunked linear attention, one block per (b,h). 64 sequential chunks of 64.
// Per chunk: scores = Q*K^T (masked); out = scores*V + Q*prefixState;
//            state += K^T*V.  State kept fp32 in LDS; final state -> d_out.
// ---------------------------------------------------------------------------
__global__ void __launch_bounds__(256) attn_chunked(const __bf16* __restrict__ Q,
                                                    const __bf16* __restrict__ Kb,
                                                    const __bf16* __restrict__ Vb,
                                                    __bf16* __restrict__ AO,
                                                    float* __restrict__ states) {
  __shared__ __bf16 sQ[CCH * DHH];
  __shared__ __bf16 sK[CCH * DHH];
  __shared__ __bf16 sV[CCH * DHH];
  __shared__ __bf16 sS[CCH * CCH];
  __shared__ float  sSt[DHH * DHH];

  int b = blockIdx.x / HHH;
  int h = blockIdx.x % HHH;
  int tid  = threadIdx.x;
  int lane = tid & 31;
  int w    = tid >> 5;
  int hh   = lane >> 4;
  int nn   = lane & 15;

  for (int i = tid; i < DHH * DHH; i += 256) sSt[i] = 0.0f;
  __syncthreads();

  int lrow = tid >> 2;
  int lc0  = (tid & 3) * 16;

  for (int n = 0; n < NCH; ++n) {
    // ---- stage chunk tiles into LDS
    size_t g = ((size_t)b * TT + (size_t)n * CCH + lrow) * DDIM + h * DHH + lc0;
    int lo = lrow * DHH + lc0;
    *(v8bf*)(&sQ[lo])     = *(const v8bf*)(Q + g);
    *(v8bf*)(&sQ[lo + 8]) = *(const v8bf*)(Q + g + 8);
    *(v8bf*)(&sK[lo])     = *(const v8bf*)(Kb + g);
    *(v8bf*)(&sK[lo + 8]) = *(const v8bf*)(Kb + g + 8);
    *(v8bf*)(&sV[lo])     = *(const v8bf*)(Vb + g);
    *(v8bf*)(&sV[lo + 8]) = *(const v8bf*)(Vb + g + 8);
    __syncthreads();

    // ---- phase 1: masked scores = Q * K^T  (64x64, 16 tiles, 2 per wave)
    for (int t = w * 2; t < w * 2 + 2; ++t) {
      int tm = t >> 2, tn = t & 3;
      v8f acc = {};
      for (int kb = 0; kb < DHH; kb += 32) {
        v16bf a  = frag_row(sQ, tm * 16, DHH, kb, lane);
        v16bf bf = frag_row(sK, tn * 16, DHH, kb, lane);
        acc = wmma_bf16(a, bf, acc);
      }
#pragma unroll
      for (int r = 0; r < 8; ++r) {
        int m = tm * 16 + r + 8 * hh;
        int c = tn * 16 + nn;
        sS[m * CCH + c] = (__bf16)((c <= m) ? acc[r] : 0.0f);
      }
    }
    __syncthreads();

    // ---- phase 2: out = S*V + Q*prefixState  (64 x DHH)
    for (int t = w * 2; t < w * 2 + 2; ++t) {
      int tm = t >> 2, tn = t & 3;
      v8f acc = {};
      for (int kb = 0; kb < CCH; kb += 32) {         // over chunk pos j
        v16bf a  = frag_row(sS, tm * 16, CCH, kb, lane);
        v16bf bf = frag_col_bf16(sV, kb, tn * 16, lane);  // B[k=j][n=d] = V[j][d]
        acc = wmma_bf16(a, bf, acc);
      }
      for (int kb = 0; kb < DHH; kb += 32) {         // over d
        v16bf a  = frag_row(sQ, tm * 16, DHH, kb, lane);
        v16bf bf = frag_col_f32(sSt, kb, tn * 16, lane);  // B[k=d][n=e] = S[d][e]
        acc = wmma_bf16(a, bf, acc);
      }
#pragma unroll
      for (int r = 0; r < 8; ++r) {
        int m = tm * 16 + r + 8 * hh;   // chunk-local row
        int c = tn * 16 + nn;           // head dim
        AO[((size_t)b * TT + (size_t)n * CCH + m) * DDIM + h * DHH + c] = (__bf16)acc[r];
      }
    }
    __syncthreads();

    // ---- phase 3: state += K^T * V  (DHH x DHH)
    for (int t = w * 2; t < w * 2 + 2; ++t) {
      int tm = t >> 2, tn = t & 3;
      v8f acc = {};
      for (int kb = 0; kb < CCH; kb += 32) {         // over chunk pos c
        v16bf a  = frag_col_bf16(sK, kb, tm * 16, lane);  // A[i][c] = K[c][i]
        v16bf bf = frag_col_bf16(sV, kb, tn * 16, lane);  // B[c][j] = V[c][j]
        acc = wmma_bf16(a, bf, acc);
      }
#pragma unroll
      for (int r = 0; r < 8; ++r)
        sSt[(tm * 16 + r + 8 * hh) * DHH + tn * 16 + nn] += acc[r];
    }
    __syncthreads();
  }

  for (int i = tid; i < DHH * DHH; i += 256)
    states[(size_t)blockIdx.x * DHH * DHH + i] = sSt[i];
}

// ---------------------------------------------------------------------------
// SwiGLU gate combine: A1 <- bf16( silu(A1) * A3 )
// ---------------------------------------------------------------------------
__global__ void silu_mul_kernel(__bf16* A1, const __bf16* __restrict__ A3) {
  size_t i = (size_t)blockIdx.x * 256 + threadIdx.x;
  float a = (float)A1[i];
  float g = a / (1.0f + __expf(-a));
  A1[i] = (__bf16)(g * (float)A3[i]);
}

// ---------------------------------------------------------------------------
// fp32 -> bf16 weight conversion
// ---------------------------------------------------------------------------
__global__ void f2bf_kernel(const float* __restrict__ src, __bf16* __restrict__ dst, int n) {
  int i = blockIdx.x * 256 + threadIdx.x;
  if (i < n) dst[i] = (__bf16)src[i];
}

// ---------------------------------------------------------------------------
// Head: logits[m,o] = sum_d Y[m,d] * bit_out_w[o,d]  (N=8, simple VALU kernel)
// ---------------------------------------------------------------------------
__global__ void head_kernel(const __bf16* __restrict__ Y, const float* __restrict__ Wout,
                            float* __restrict__ logits) {
  size_t idx = (size_t)blockIdx.x * 256 + threadIdx.x;   // over MTOT*8
  int o = (int)(idx & 7);
  size_t m = idx >> 3;
  const __bf16* y = Y + m * DDIM;
  const float* ww = Wout + (size_t)o * DDIM;
  float s = 0.0f;
  for (int d = 0; d < DDIM; ++d) s += (float)y[d] * ww[d];
  logits[idx] = s;
}

// ---------------------------------------------------------------------------
extern "C" void kernel_launch(void* const* d_in, const int* in_sizes, int n_in,
                              void* d_out, int out_size, void* d_ws, size_t ws_size,
                              hipStream_t stream) {
  (void)in_sizes; (void)n_in; (void)out_size; (void)ws_size;

  const int*   ids  = (const int*)d_in[0];
  const float* bitw = (const float*)d_in[1];
  const float* pos  = (const float*)d_in[2];
  const float* n1w  = (const float*)d_in[3];
  const float* wq   = (const float*)d_in[4];
  const float* wk   = (const float*)d_in[5];
  const float* wv   = (const float*)d_in[6];
  const float* wo   = (const float*)d_in[7];
  const float* n2w  = (const float*)d_in[8];
  const float* w1   = (const float*)d_in[9];
  const float* w3   = (const float*)d_in[10];
  const float* w2   = (const float*)d_in[11];
  const float* fnw  = (const float*)d_in[12];
  const float* bow  = (const float*)d_in[13];

  float* logits = (float*)d_out;
  float* states = logits + (size_t)MTOT * 8;   // (L,B,H,DH,DH) fp32

  char* ws = (char*)d_ws;
  size_t off = 0;
  float*  X   = (float*) (ws + off); off += (size_t)MTOT * DDIM * 4;
  __bf16* Y   = (__bf16*)(ws + off); off += (size_t)MTOT * DDIM * 2;
  __bf16* Qb  = (__bf16*)(ws + off); off += (size_t)MTOT * DDIM * 2;
  __bf16* Kb  = (__bf16*)(ws + off); off += (size_t)MTOT * DDIM * 2;
  __bf16* Vb  = (__bf16*)(ws + off); off += (size_t)MTOT * DDIM * 2;
  __bf16* AOb = (__bf16*)(ws + off); off += (size_t)MTOT * DDIM * 2;
  __bf16* A1  = (__bf16*)(ws + off); off += (size_t)MTOT * DFFF * 2;
  __bf16* A3  = (__bf16*)(ws + off); off += (size_t)MTOT * DFFF * 2;
  __bf16* Wqb = (__bf16*)(ws + off); off += (size_t)DDIM * DDIM * 2;
  __bf16* Wkb = (__bf16*)(ws + off); off += (size_t)DDIM * DDIM * 2;
  __bf16* Wvb = (__bf16*)(ws + off); off += (size_t)DDIM * DDIM * 2;
  __bf16* Wob = (__bf16*)(ws + off); off += (size_t)DDIM * DDIM * 2;
  __bf16* W1b = (__bf16*)(ws + off); off += (size_t)DFFF * DDIM * 2;
  __bf16* W3b = (__bf16*)(ws + off); off += (size_t)DFFF * DDIM * 2;
  __bf16* W2b = (__bf16*)(ws + off); off += (size_t)DDIM * DFFF * 2;

  dim3 blk(256);
  dim3 gD  (DDIM / 64,  MTOT / 128);   // N=1024 GEMMs
  dim3 gFF (DFFF / 64,  MTOT / 128);   // N=2816 GEMMs

  embed_kernel<<<(MTOT * DDIM) / 256, blk, 0, stream>>>(ids, bitw, pos, X);

  const int nDD = DDIM * DDIM, nFD = DFFF * DDIM;
  for (int l = 0; l < LLL; ++l) {
    f2bf_kernel<<<nDD / 256, blk, 0, stream>>>(wq + (size_t)l * nDD, Wqb, nDD);
    f2bf_kernel<<<nDD / 256, blk, 0, stream>>>(wk + (size_t)l * nDD, Wkb, nDD);
    f2bf_kernel<<<nDD / 256, blk, 0, stream>>>(wv + (size_t)l * nDD, Wvb, nDD);
    f2bf_kernel<<<nDD / 256, blk, 0, stream>>>(wo + (size_t)l * nDD, Wob, nDD);
    f2bf_kernel<<<nFD / 256, blk, 0, stream>>>(w1 + (size_t)l * nFD, W1b, nFD);
    f2bf_kernel<<<nFD / 256, blk, 0, stream>>>(w3 + (size_t)l * nFD, W3b, nFD);
    f2bf_kernel<<<nFD / 256, blk, 0, stream>>>(w2 + (size_t)l * nFD, W2b, nFD);

    rmsnorm_kernel<<<MTOT, blk, 0, stream>>>(X, n1w + (size_t)l * DDIM, Y);
    gemm_bf16_wmma<EPI_SCALE><<<gD, blk, 0, stream>>>(Y, Wqb, nullptr, nullptr, Qb,
                                                      MTOT, DDIM, DDIM, SCALEF);
    gemm_bf16_wmma<EPI_BIN>  <<<gD, blk, 0, stream>>>(Y, Wkb, nullptr, nullptr, Kb,
                                                      MTOT, DDIM, DDIM, 1.0f);
    gemm_bf16_wmma<EPI_BIN>  <<<gD, blk, 0, stream>>>(Y, Wvb, nullptr, nullptr, Vb,
                                                      MTOT, DDIM, DDIM, 1.0f);
    attn_chunked<<<BB * HHH, blk, 0, stream>>>(Qb, Kb, Vb, AOb,
                                               states + (size_t)l * BB * HHH * DHH * DHH);
    gemm_bf16_wmma<EPI_RESID><<<gD, blk, 0, stream>>>(AOb, Wob, X, X, nullptr,
                                                      MTOT, DDIM, DDIM, 1.0f);

    rmsnorm_kernel<<<MTOT, blk, 0, stream>>>(X, n2w + (size_t)l * DDIM, Y);
    gemm_bf16_wmma<EPI_BF16><<<gFF, blk, 0, stream>>>(Y, W1b, nullptr, nullptr, A1,
                                                      MTOT, DFFF, DDIM, 1.0f);
    gemm_bf16_wmma<EPI_BF16><<<gFF, blk, 0, stream>>>(Y, W3b, nullptr, nullptr, A3,
                                                      MTOT, DFFF, DDIM, 1.0f);
    silu_mul_kernel<<<(MTOT * DFFF) / 256, blk, 0, stream>>>(A1, A3);
    gemm_bf16_wmma<EPI_RESID><<<gD, blk, 0, stream>>>(A1, W2b, X, X, nullptr,
                                                      MTOT, DDIM, DFFF, 1.0f);
  }

  rmsnorm_kernel<<<MTOT, blk, 0, stream>>>(X, fnw, Y);
  head_kernel<<<(MTOT * 8) / 256, blk, 0, stream>>>(Y, bow, logits);
}
